// VectorQuantizer_56581899157662
// MI455X (gfx1250) — matrix operand order repaired and checked by reading, not compiled, not run on hardware
//
#include <hip/hip_runtime.h>
#include <hip/hip_bf16.h>

typedef __bf16 bf16_t;
typedef __attribute__((ext_vector_type(16))) __bf16 v16bf;
typedef __attribute__((ext_vector_type(8)))  __bf16 v8bf;
typedef __attribute__((ext_vector_type(8)))  float  v8f;

#define NUM_EMB 8192
#define EMB_D   512
#define BB      8
#define LL      4096
#define N_ROWS  (BB * LL)                  // 32768
#define TOT_OUT ((size_t)BB * EMB_D * LL)  // 16777216
#define JSTEPS  (EMB_D / 32)               // 16

// ---------------------------------------------------------------------------
// WMMA helper:  D = A(16x32 bf16) * B(32x16 bf16) + C(16x16 f32)
// ---------------------------------------------------------------------------
__device__ __forceinline__ v8f wmma_bf16(v16bf a, v16bf b, v8f c) {
  return __builtin_amdgcn_wmma_f32_16x16x32_bf16(
      /*neg_a=*/false, a, /*neg_b=*/false, b,
      /*c_mod=*/(short)0, c, /*reuse_a=*/false, /*reuse_b=*/false);
}

union FragU { v16bf v; v8bf h[2]; };

// A fragment (16x32, M x K). Row-major source [rows][EMB_D].
// Lane l holds M = l&15. lanes<16: K = k0..k0+15;
// lanes>=16: K = k0+16..k0+31.
__device__ __forceinline__ v16bf load_frag_a(const bf16_t* __restrict__ base,
                                             int row0, int k0, int lane) {
  int row = row0 + (lane & 15);
  int k   = k0 + ((lane >> 4) << 4);           // +0 or +16
  const bf16_t* p = base + (size_t)row * EMB_D + k;
  FragU u;
  u.h[0] = *(const v8bf*)(p);                  // K = k .. k+7
  u.h[1] = *(const v8bf*)(p + 8);              // K = k+8 .. k+15
  return u.v;
}

// B fragment (32x16, K x N), column n = source row (row-major [rows][EMB_D]).
// Lane l holds N = l&15; lanes<16: K = k0..k0+15, lanes>=16: K = k0+16..k0+31.
__device__ __forceinline__ v16bf load_frag_b(const bf16_t* __restrict__ base,
                                             int col0, int k0, int lane) {
  int row = col0 + (lane & 15);
  int k   = k0 + ((lane >> 4) << 4);           // +0 or +16
  const bf16_t* p = base + (size_t)row * EMB_D + k;
  FragU u;
  u.h[0] = *(const v8bf*)(p);
  u.h[1] = *(const v8bf*)(p + 8);
  return u.v;
}

// CDNA5 async copy: 16B global -> LDS, tracked by ASYNCcnt.
// LDS byte address = low 32 bits of the generic shared-aperture VA.
__device__ __forceinline__ void async_copy_b128(const bf16_t* gsrc, bf16_t* lds_dst) {
  unsigned laddr = (unsigned)(size_t)lds_dst;
  asm volatile("global_load_async_to_lds_b128 %0, %1, off"
               :: "v"(laddr), "v"(gsrc) : "memory");
}

__device__ __forceinline__ void wait_asynccnt_le2() {
#if __has_builtin(__builtin_amdgcn_s_wait_asynccnt)
  __builtin_amdgcn_s_wait_asynccnt(2);
#else
  asm volatile("s_wait_asynccnt 0x2" ::: "memory");
#endif
}
__device__ __forceinline__ void wait_asynccnt_0() {
#if __has_builtin(__builtin_amdgcn_s_wait_asynccnt)
  __builtin_amdgcn_s_wait_asynccnt(0);
#else
  asm volatile("s_wait_asynccnt 0x0" ::: "memory");
#endif
}

// monotone float -> uint map for unsigned-min ordering
__device__ __forceinline__ unsigned fmap(float f) {
  unsigned u = __float_as_uint(f);
  return (u & 0x80000000u) ? ~u : (u | 0x80000000u);
}

// ---------------------------------------------------------------------------
// 1) transpose [B, D, L] f32 -> x_bf16 [B*L, D]
// ---------------------------------------------------------------------------
__global__ void k_transpose_x(const float* __restrict__ in, bf16_t* __restrict__ xb) {
  __shared__ bf16_t tile[32][34];
  int b  = blockIdx.z;
  int l0 = blockIdx.x * 32;
  int d0 = blockIdx.y * 32;
  int tx = threadIdx.x, ty = threadIdx.y;      // 32 x 8
  #pragma unroll
  for (int i = 0; i < 32; i += 8) {
    int d = d0 + ty + i;
    tile[ty + i][tx] = (bf16_t)in[(size_t)b * EMB_D * LL + (size_t)d * LL + l0 + tx];
  }
  __syncthreads();
  #pragma unroll
  for (int i = 0; i < 32; i += 8) {
    int l = l0 + ty + i;
    xb[(size_t)(b * LL + l) * EMB_D + d0 + tx] = tile[tx][ty + i];
  }
}

// 2) split f32 -> (hi, lo) bf16 pair
__global__ void k_split_hilo(const float* __restrict__ in, bf16_t* __restrict__ hi,
                             bf16_t* __restrict__ lo, int n) {
  int i = blockIdx.x * 256 + threadIdx.x;
  if (i < n) {
    float v = in[i];
    bf16_t h = (bf16_t)v;
    hi[i] = h;
    lo[i] = (bf16_t)(v - (float)h);
  }
}

// ---------------------------------------------------------------------------
// 3) GEMM1: codebook[k][d] = sum_j emb[k][j] * proj[d][j]  (split-bf16 ~ fp32)
// ---------------------------------------------------------------------------
__global__ void __launch_bounds__(256) k_gemm_codebook(
    const bf16_t* __restrict__ ehi, const bf16_t* __restrict__ elo,
    const bf16_t* __restrict__ phi, const bf16_t* __restrict__ plo,
    float* __restrict__ cb) {
  int lane = threadIdx.x & 31;
  int wave = threadIdx.x >> 5;
  int m0 = blockIdx.x * 128 + wave * 16;       // code rows
  int n0 = blockIdx.y * 128;                   // output-d columns
  v8f acc[8];
  #pragma unroll
  for (int c = 0; c < 8; ++c) acc[c] = (v8f){0.f,0.f,0.f,0.f,0.f,0.f,0.f,0.f};

  for (int j = 0; j < EMB_D; j += 32) {
    v16bf ah = load_frag_a(ehi, m0, j, lane);
    v16bf al = load_frag_a(elo, m0, j, lane);
    #pragma unroll
    for (int c = 0; c < 8; ++c) {
      v16bf bh = load_frag_b(phi, n0 + c * 16, j, lane);
      v16bf bl = load_frag_b(plo, n0 + c * 16, j, lane);
      acc[c] = wmma_bf16(ah, bh, acc[c]);
      acc[c] = wmma_bf16(ah, bl, acc[c]);
      acc[c] = wmma_bf16(al, bh, acc[c]);
    }
  }
  #pragma unroll
  for (int c = 0; c < 8; ++c) {
    #pragma unroll
    for (int r = 0; r < 8; ++r) {
      int m = m0 + r + ((lane >> 4) << 3);     // VGPR r: M=r (lo half), r+8 (hi half)
      int n = n0 + c * 16 + (lane & 15);
      cb[(size_t)m * EMB_D + n] = acc[c][r];
    }
  }
}

// 4) per-code squared norm + bf16 copy of codebook
__global__ void k_cb_norm(const float* __restrict__ cb, bf16_t* __restrict__ cbb,
                          float* __restrict__ cnorm) {
  int k = blockIdx.x;
  int t = threadIdx.x;
  float s = 0.f;
  for (int d = t; d < EMB_D; d += 256) {
    float v = cb[(size_t)k * EMB_D + d];
    cbb[(size_t)k * EMB_D + d] = (bf16_t)v;
    s += v * v;
  }
  __shared__ float red[256];
  red[t] = s; __syncthreads();
  for (int o = 128; o > 0; o >>= 1) { if (t < o) red[t] += red[t + o]; __syncthreads(); }
  if (t == 0) cnorm[k] = red[0];
}

// 5) init packed argmin state
__global__ void k_init_best(unsigned long long* best) {
  int i = blockIdx.x * 256 + threadIdx.x;
  if (i < N_ROWS) best[i] = ~0ull;
}

// ---------------------------------------------------------------------------
// 6) GEMM2 + fused argmin. Codebook tile is staged into LDS with CDNA5
//    GLOBAL_LOAD_ASYNC_TO_LDS_B128 (ASYNCcnt), double-buffered and overlapped
//    with WMMA compute. score = |c|^2 - 2 x.c ; packed atomicMin per row.
//    block = 8 waves, tile 128(rows) x 128(codes).
// ---------------------------------------------------------------------------
#define BROW 40   // LDS row stride in bf16 (80B) to spread banks

__global__ void __launch_bounds__(256) k_gemm_argmin(
    const bf16_t* __restrict__ xb, const bf16_t* __restrict__ cbb,
    const float* __restrict__ cnorm, unsigned long long* __restrict__ best) {
  __shared__ bf16_t btile[2][128 * BROW];      // 2 x 10KB
  int lane = threadIdx.x & 31;
  int wave = threadIdx.x >> 5;
  int tid  = threadIdx.x;
  int m0 = blockIdx.y * 128 + wave * 16;       // token rows
  int k0 = blockIdx.x * 128;                   // code columns

  // stage codes [k0..k0+127], d [j..j+31] into buffer `buf`:
  // 512 x 16B chunks, 2 async b128 per thread
  auto stage = [&](int buf, int j) {
    #pragma unroll
    for (int t = tid; t < 512; t += 256) {
      int code = t >> 2;                       // 0..127
      int part = t & 3;                        // 16B chunk within 64B row
      const bf16_t* g = cbb + (size_t)(k0 + code) * EMB_D + j + part * 8;
      bf16_t* l = &btile[buf][code * BROW + part * 8];
      async_copy_b128(g, l);
    }
  };

  stage(0, 0);

  v8f acc[8];
  #pragma unroll
  for (int c = 0; c < 8; ++c) acc[c] = (v8f){0.f,0.f,0.f,0.f,0.f,0.f,0.f,0.f};

  for (int js = 0; js < JSTEPS; ++js) {
    int j = js * 32;
    if (js + 1 < JSTEPS) {
      stage((js + 1) & 1, j + 32);             // prefetch next tile (other buffer)
      wait_asynccnt_le2();                     // in-order: current tile complete
    } else {
      wait_asynccnt_0();
    }
    __syncthreads();                           // all waves' copies visible

    v16bf a = load_frag_a(xb, m0, j, lane);
    const bf16_t* bt = &btile[js & 1][0];
    #pragma unroll
    for (int c = 0; c < 8; ++c) {
      int code = c * 16 + (lane & 15);
      int kk   = (lane >> 4) << 4;             // 0 or 16
      const bf16_t* p = bt + code * BROW + kk;
      FragU u;
      u.h[0] = *(const v8bf*)(p);              // ds_load_b128
      u.h[1] = *(const v8bf*)(p + 8);
      acc[c] = wmma_bf16(a, u.v, acc[c]);
    }
    __syncthreads();                           // reads done before next overwrite
  }

  float cn[8];
  #pragma unroll
  for (int c = 0; c < 8; ++c) cn[c] = cnorm[k0 + c * 16 + (lane & 15)];

  #pragma unroll
  for (int r = 0; r < 8; ++r) {
    float bs = cn[0] - 2.f * acc[0][r];
    int   bc = k0 + (lane & 15);
    #pragma unroll
    for (int c = 1; c < 8; ++c) {
      float s = cn[c] - 2.f * acc[c][r];
      int code = k0 + c * 16 + (lane & 15);
      if (s < bs) { bs = s; bc = code; }
    }
    unsigned long long key =
        ((unsigned long long)fmap(bs) << 32) | (unsigned)bc;
    #pragma unroll
    for (int off = 1; off < 16; off <<= 1) {
      unsigned long long o = __shfl_xor(key, off, 32);
      if (o < key) key = o;
    }
    if ((lane & 15) == 0) {
      int m = m0 + r + ((lane >> 4) << 3);
      atomicMin(&best[m], key);
    }
  }
}

// ---------------------------------------------------------------------------
// 7) gather + transpose-out + per-block loss partials + index output
// ---------------------------------------------------------------------------
__global__ void k_finalize(const float* __restrict__ in, const float* __restrict__ cb,
                           const unsigned long long* __restrict__ best,
                           float* __restrict__ out, float* __restrict__ out_idx,
                           float* __restrict__ partials) {
  size_t g = (size_t)blockIdx.x * 256 + threadIdx.x;  // enumerates (b, d, l)
  int l = (int)(g & (LL - 1));
  int d = (int)((g >> 12) & (EMB_D - 1));
  int b = (int)(g >> 21);
  int n = b * LL + l;
  unsigned idx = (unsigned)(best[n] & 0xffffffffull);
  float q = cb[(size_t)idx * EMB_D + d];
  float x = in[g];
  out[g] = q;                                  // straight-through fwd value
  if (d == 0) out_idx[n] = (float)idx;
  float diff = q - x;
  __shared__ float red[256];
  red[threadIdx.x] = diff * diff;
  __syncthreads();
  for (int o = 128; o > 0; o >>= 1) {
    if (threadIdx.x < o) red[threadIdx.x] += red[threadIdx.x + o];
    __syncthreads();
  }
  if (threadIdx.x == 0) partials[blockIdx.x] = red[0];
}

__global__ void k_loss_reduce(const float* __restrict__ partials, float* __restrict__ loss_out,
                              int np) {
  float s = 0.f;
  for (int i = threadIdx.x; i < np; i += 256) s += partials[i];
  __shared__ float red[256];
  red[threadIdx.x] = s; __syncthreads();
  for (int o = 128; o > 0; o >>= 1) {
    if (threadIdx.x < o) red[threadIdx.x] += red[threadIdx.x + o];
    __syncthreads();
  }
  // loss = q_latent + 0.25 * e_latent; both equal mean((q-x)^2) in fwd
  if (threadIdx.x == 0) *loss_out = 1.25f * red[0] / (float)TOT_OUT;
}

// ---------------------------------------------------------------------------
extern "C" void kernel_launch(void* const* d_in, const int* in_sizes, int n_in,
                              void* d_out, int out_size, void* d_ws, size_t ws_size,
                              hipStream_t stream) {
  const float* inputs = (const float*)d_in[0];   // [B, D, L]
  const float* emb    = (const float*)d_in[1];   // [K, D]
  const float* proj   = (const float*)d_in[2];   // [D, D]
  float* out = (float*)d_out;

  // workspace carving (256B aligned)
  char* ws = (char*)d_ws;
  size_t off = 0;
  auto carve = [&](size_t bytes) -> char* {
    char* p = ws + off;
    off += (bytes + 255) & ~(size_t)255;
    return p;
  };
  bf16_t* xb      = (bf16_t*)carve((size_t)N_ROWS * EMB_D * 2);   // 32 MB
  bf16_t* ehi     = (bf16_t*)carve((size_t)NUM_EMB * EMB_D * 2);  // 8 MB
  bf16_t* elo     = (bf16_t*)carve((size_t)NUM_EMB * EMB_D * 2);  // 8 MB
  bf16_t* phi     = (bf16_t*)carve((size_t)EMB_D * EMB_D * 2);    // 0.5 MB
  bf16_t* plo     = (bf16_t*)carve((size_t)EMB_D * EMB_D * 2);    // 0.5 MB
  float*  cb      = (float*) carve((size_t)NUM_EMB * EMB_D * 4);  // 16 MB
  bf16_t* cbb     = (bf16_t*)carve((size_t)NUM_EMB * EMB_D * 2);  // 8 MB
  float*  cnorm   = (float*) carve((size_t)NUM_EMB * 4);
  unsigned long long* best = (unsigned long long*)carve((size_t)N_ROWS * 8);
  const int n_fin_blocks = (int)(TOT_OUT / 256);                  // 65536
  float* partials = (float*)carve((size_t)n_fin_blocks * 4);

  float* loss_out = out + TOT_OUT;
  float* out_idx  = out + TOT_OUT + 1;

  // 1) x -> bf16, transposed to [N, D]
  k_transpose_x<<<dim3(LL / 32, EMB_D / 32, BB), dim3(32, 8), 0, stream>>>(inputs, xb);
  // 2) hi/lo splits
  k_split_hilo<<<(NUM_EMB * EMB_D + 255) / 256, 256, 0, stream>>>(emb, ehi, elo, NUM_EMB * EMB_D);
  k_split_hilo<<<(EMB_D * EMB_D + 255) / 256, 256, 0, stream>>>(proj, phi, plo, EMB_D * EMB_D);
  // 3) codebook GEMM (split-bf16 ~ fp32)
  k_gemm_codebook<<<dim3(NUM_EMB / 128, EMB_D / 128), 256, 0, stream>>>(ehi, elo, phi, plo, cb);
  // 4) norms + bf16 codebook
  k_cb_norm<<<NUM_EMB, 256, 0, stream>>>(cb, cbb, cnorm);
  // 5) argmin state
  k_init_best<<<N_ROWS / 256, 256, 0, stream>>>(best);
  // 6) distance GEMM (async-LDS staged) + fused argmin
  k_gemm_argmin<<<dim3(NUM_EMB / 128, N_ROWS / 128), 256, 0, stream>>>(xb, cbb, cnorm, best);
  // 7) gather / transpose-out / loss partials / indices
  k_finalize<<<n_fin_blocks, 256, 0, stream>>>(inputs, cb, best, out, out_idx, partials);
  // 8) deterministic loss reduction
  k_loss_reduce<<<1, 256, 0, stream>>>(partials, loss_out, n_fin_blocks);
}
